// CausalSelfAttention_70437463654402
// MI455X (gfx1250) — matrix operand order repaired and checked by reading, not compiled, not run on hardware
//
#include <hip/hip_runtime.h>
#include <stdint.h>

typedef unsigned int  u32;
typedef unsigned short u16;
typedef __bf16 v16bf __attribute__((ext_vector_type(16)));
typedef float  v8f   __attribute__((ext_vector_type(8)));

union Frag16 { v16bf v; uint4 q[2]; u32 u[8]; };

__device__ __forceinline__ u16 f2bf(float f) {
  u32 u = __float_as_uint(f);
  u32 r = u + 0x7FFFu + ((u >> 16) & 1u);   // round-to-nearest-even
  return (u16)(r >> 16);
}
__device__ __forceinline__ float bf2f(u16 h) {
  return __uint_as_float(((u32)h) << 16);
}
__device__ __forceinline__ v8f wmma_bf16(const Frag16& a, const Frag16& b, v8f c) {
  return __builtin_amdgcn_wmma_f32_16x16x32_bf16(false, a.v, false, b.v, (short)0, c, false, false);
}

// CDNA5 async DMA: 16B global -> LDS, tracked with ASYNCcnt (ISA 15.18.3 op 98)
__device__ __forceinline__ void async_ld_b128(u32 lds_off, const u16* g) {
  asm volatile("global_load_async_to_lds_b128 %0, %1, off"
               :: "v"(lds_off), "v"((unsigned long long)(size_t)g)
               : "memory");
}
__device__ __forceinline__ u32 lds_off_of(const void* p) {
  return (u32)(size_t)p;   // low 32 bits of flat LDS address = LDS byte offset
}

// ---------------- fp32 -> bf16 conversion ----------------
__global__ void cvt_f32_bf16(const float* __restrict__ in, u16* __restrict__ out, int n) {
  int i = blockIdx.x * blockDim.x + threadIdx.x;
  if (i < n) out[i] = f2bf(in[i]);
}

// ---------------- fp32 [K][N] -> bf16 transposed [N][K] ----------------
__global__ void cvt_transpose_bf16(const float* __restrict__ in, u16* __restrict__ out,
                                   int K, int N) {
  int i = blockIdx.x * blockDim.x + threadIdx.x;
  if (i < K * N) {
    int k = i / N, n = i - k * N;
    out[(size_t)n * K + k] = f2bf(in[i]);
  }
}

// ---------------- RoPE (in-place on bf16 [B,H,T,HD]) ----------------
__global__ void rope_kernel(u16* __restrict__ buf,
                            const float* __restrict__ cosp,
                            const float* __restrict__ sinp,
                            float scale, int npairs) {
  int i = blockIdx.x * blockDim.x + threadIdx.x;  // pair index over B*H*T*64
  if (i >= npairs) return;
  int p = i & 63;
  int t = (i >> 6) & 2047;
  size_t off = ((size_t)(i >> 6)) * 128 + 2 * p;
  float c = cosp[t * 64 + p], s = sinp[t * 64 + p];
  float x0 = bf2f(buf[off]), x1 = bf2f(buf[off + 1]);
  buf[off]     = f2bf((x0 * c - x1 * s) * scale);
  buf[off + 1] = f2bf((x0 * s + x1 * c) * scale);
}

// ---------------- bf16 GEMM: C = A[MxK] * Bt[NxK]^T + bias ----------------
// 128x128 block tile, 8 waves (4Mx2N), each wave 32x64 = 8 WMMAs / K-step.
// Tiles staged with global_load_async_to_lds_b128, double-buffered.
// mode 0: scatter bf16 to Q[B,H,T,HD], K[B,H,T,HD], V transposed [B,H,HD,T]
// mode 1: write f32 to outp[M x N]
#define LSTR 40   // padded LDS row stride (80B = 20 banks -> conflict-free b128)

__global__ __launch_bounds__(256)
void gemm_bf16_kernel(const u16* __restrict__ A, const u16* __restrict__ Bt,
                      const float* __restrict__ bias, int M, int N, int K,
                      int mode, u16* __restrict__ qd, u16* __restrict__ kd,
                      u16* __restrict__ vd, float* __restrict__ outp)
{
  __shared__ u16 As[2][128 * LSTR];
  __shared__ u16 Bs[2][128 * LSTR];   // B^T tile: [n][k]
  const int tid = threadIdx.x;
  const int lane = tid & 31, wid = tid >> 5;
  const int half = lane >> 4, l16 = lane & 15;
  const int wm = wid & 3, wn = wid >> 2;
  const int bm = blockIdx.y * 128;
  const int bn = blockIdx.x * 128;

  v8f acc[2][4];
  for (int i = 0; i < 2; i++)
    for (int j = 0; j < 4; j++)
      for (int r = 0; r < 8; r++) acc[i][j][r] = 0.f;

  // per-thread async chunk assignment: 512 16B chunks per tile, 2 per thread
  const int c0 = tid * 2;

  auto issue_tile = [&](int buf, int k0) {
#pragma unroll
    for (int s = 0; s < 2; s++) {
      int c = c0 + s;
      int row = c >> 2, kc = (c & 3) * 8;
      async_ld_b128(lds_off_of(&As[buf][row * LSTR + kc]),
                    A + (size_t)(bm + row) * K + k0 + kc);
      async_ld_b128(lds_off_of(&Bs[buf][row * LSTR + kc]),
                    Bt + (size_t)(bn + row) * K + k0 + kc);
    }
  };

  issue_tile(0, 0);
  const int nk = K >> 5;
  for (int it = 0; it < nk; ++it) {
    const int cur = it & 1;
    if (it + 1 < nk) {
      issue_tile(cur ^ 1, (it + 1) << 5);
      asm volatile("s_wait_asynccnt 0x4" ::: "memory");  // cur tile landed, next in flight
    } else {
      asm volatile("s_wait_asynccnt 0x0" ::: "memory");
    }
    __syncthreads();

    Frag16 af[2], bfr[4];
#pragma unroll
    for (int i = 0; i < 2; i++) {
      int m = wm * 32 + i * 16 + l16;
      af[i].q[0] = *(const uint4*)&As[cur][m * LSTR + 16 * half];
      af[i].q[1] = *(const uint4*)&As[cur][m * LSTR + 16 * half + 8];
    }
#pragma unroll
    for (int j = 0; j < 4; j++) {
      int n = wn * 64 + j * 16 + l16;
      bfr[j].q[0] = *(const uint4*)&Bs[cur][n * LSTR + 16 * half];
      bfr[j].q[1] = *(const uint4*)&Bs[cur][n * LSTR + 16 * half + 8];
    }
#pragma unroll
    for (int i = 0; i < 2; i++)
#pragma unroll
      for (int j = 0; j < 4; j++)
        acc[i][j] = wmma_bf16(af[i], bfr[j], acc[i][j]);
    __syncthreads();   // all waves done reading before next overwrite
  }

#pragma unroll
  for (int i = 0; i < 2; i++)
#pragma unroll
    for (int j = 0; j < 4; j++) {
      int col = bn + wn * 64 + j * 16 + l16;
      float bv = bias[col];
#pragma unroll
      for (int r = 0; r < 8; r++) {
        int row = bm + wm * 32 + i * 16 + r + 8 * half;
        float val = acc[i][j][r] + bv;
        if (mode == 0) {
          int mat = col >> 11, rem = col & 2047;
          int h = rem >> 7, d = rem & 127;
          int b = row >> 11, t = row & 2047;
          if (mat == 2) {     // V stored transposed per head: [B,H,HD,T]
            vd[(((size_t)(b * 16 + h)) * 128 + d) * 2048 + t] = f2bf(val);
          } else {
            u16* dst = (mat == 0) ? qd : kd;
            dst[(((size_t)(b * 16 + h)) * 2048 + t) * 128 + d] = f2bf(val);
          }
        } else {
          outp[(size_t)row * N + col] = val;
        }
      }
    }
}

// ---------------- flash attention (one wave = 16 query rows) ----------------
// V is pre-transposed [B,H,HD,T], so both K^T and V B-fragments are pure b128 loads.
__global__ __launch_bounds__(128)
void flash_attn_kernel(const u16* __restrict__ Q, const u16* __restrict__ Kb,
                       const u16* __restrict__ Vt, u16* __restrict__ Y)
{
  __shared__ u16 pls[4][16 * LSTR];
  const int tid = threadIdx.x, lane = tid & 31, wid = tid >> 5;
  const int half = lane >> 4, l16 = lane & 15;
  const int gw = blockIdx.x * 4 + wid;       // 0..4095
  const int bh = gw >> 7;                    // 0..31
  const int qt = gw & 127;
  const int b = bh >> 4, h = bh & 15;
  const int q0 = qt * 16;
  const u16* Qp = Q  + (size_t)bh * 2048 * 128;
  const u16* Kp = Kb + (size_t)bh * 2048 * 128;
  const u16* Vp = Vt + (size_t)bh * 128 * 2048;   // [d][t]

  // Q fragments (A-layout), 4 chunks of K=32 covering HD=128
  Frag16 qf[4];
#pragma unroll
  for (int dc = 0; dc < 4; dc++) {
    const u16* base = Qp + (size_t)(q0 + l16) * 128 + dc * 32;
    qf[dc].q[0] = *(const uint4*)(base + 16 * half);
    qf[dc].q[1] = *(const uint4*)(base + 16 * half + 8);
  }

  v8f o[8];
  for (int c = 0; c < 8; c++)
    for (int r = 0; r < 8; r++) o[c][r] = 0.f;
  float mrow[8], lrow[8];
  for (int r = 0; r < 8; r++) { mrow[r] = -3e38f; lrow[r] = 0.f; }

  const int kend = q0 + 15;
  for (int kb = 0; kb <= kend; kb += 32) {
    v8f s0, s1;
    for (int r = 0; r < 8; r++) { s0[r] = 0.f; s1[r] = 0.f; }
    // S = Q * K^T : B-fragment of K^T == contiguous rows of K
#pragma unroll
    for (int dc = 0; dc < 4; dc++) {
      Frag16 kf0, kf1;
      const u16* k0p = Kp + (size_t)(kb + l16) * 128 + dc * 32 + 16 * half;
      kf0.q[0] = *(const uint4*)(k0p);
      kf0.q[1] = *(const uint4*)(k0p + 8);
      const u16* k1p = Kp + (size_t)(kb + 16 + l16) * 128 + dc * 32 + 16 * half;
      kf1.q[0] = *(const uint4*)(k1p);
      kf1.q[1] = *(const uint4*)(k1p + 8);
      s0 = wmma_bf16(qf[dc], kf0, s0);
      s1 = wmma_bf16(qf[dc], kf1, s1);
    }
    // causal mask (diagonal blocks only)
    if (kb + 31 > q0) {
      int key0 = kb + l16, key1 = kb + 16 + l16;
#pragma unroll
      for (int r = 0; r < 8; r++) {
        int qrow = q0 + r + 8 * half;
        if (key0 > qrow) s0[r] = -3e38f;
        if (key1 > qrow) s1[r] = -3e38f;
      }
    }
    // online softmax in C-register layout; row reductions across 16 lanes
#pragma unroll
    for (int r = 0; r < 8; r++) {
      float t = fmaxf(s0[r], s1[r]);
      t = fmaxf(t, __shfl_xor(t, 1, 16));
      t = fmaxf(t, __shfl_xor(t, 2, 16));
      t = fmaxf(t, __shfl_xor(t, 4, 16));
      t = fmaxf(t, __shfl_xor(t, 8, 16));
      float mn = fmaxf(mrow[r], t);
      float p0 = __expf(s0[r] - mn);
      float p1 = __expf(s1[r] - mn);
      float rs = p0 + p1;
      rs += __shfl_xor(rs, 1, 16);
      rs += __shfl_xor(rs, 2, 16);
      rs += __shfl_xor(rs, 4, 16);
      rs += __shfl_xor(rs, 8, 16);
      float sc = __expf(mrow[r] - mn);
      lrow[r] = lrow[r] * sc + rs;
      mrow[r] = mn;
#pragma unroll
      for (int c = 0; c < 8; c++) o[c][r] *= sc;
      int m = r + 8 * half;                       // C-layout -> LDS (16x32 P tile)
      pls[wid][m * LSTR + l16]      = f2bf(p0);
      pls[wid][m * LSTR + 16 + l16] = f2bf(p1);
    }
    // reload P in A-layout, accumulate O += P * V  (V fragments: b128 from Vt)
    Frag16 pf;
    pf.q[0] = *(const uint4*)&pls[wid][l16 * LSTR + 16 * half];
    pf.q[1] = *(const uint4*)&pls[wid][l16 * LSTR + 16 * half + 8];
#pragma unroll
    for (int c = 0; c < 8; c++) {
      Frag16 vf;
      const u16* v0p = Vp + (size_t)(c * 16 + l16) * 2048 + kb + 16 * half;
      vf.q[0] = *(const uint4*)(v0p);
      vf.q[1] = *(const uint4*)(v0p + 8);
      o[c] = wmma_bf16(pf, vf, o[c]);
    }
  }

  // normalize and write bf16 y [B*T, C] row-major (row = b*T+t, col = h*128+d)
#pragma unroll
  for (int r = 0; r < 8; r++) {
    float inv = 1.0f / lrow[r];
    int row = b * 2048 + q0 + r + 8 * half;
#pragma unroll
    for (int c = 0; c < 8; c++) {
      int col = h * 128 + c * 16 + l16;
      Y[(size_t)row * 2048 + col] = f2bf(o[c][r] * inv);
    }
  }
}

// ---------------- host launcher ----------------
extern "C" void kernel_launch(void* const* d_in, const int* in_sizes, int n_in,
                              void* d_out, int out_size, void* d_ws, size_t ws_size,
                              hipStream_t stream) {
  const float* x      = (const float*)d_in[0];   // [2,2048,2048]
  const float* W_attn = (const float*)d_in[1];   // [2048,6144]
  const float* b_attn = (const float*)d_in[2];   // [6144]
  const float* W_proj = (const float*)d_in[3];   // [2048,2048]
  const float* b_proj = (const float*)d_in[4];   // [2048]
  const float* cosp   = (const float*)d_in[5];   // [2048,64]
  const float* sinp   = (const float*)d_in[6];   // [2048,64]
  float* out = (float*)d_out;

  const int M = 4096, C = 2048, N3 = 6144;
  const int NX  = M * C;                 // 8,388,608
  const int NWA = C * N3;                // 12,582,912
  const int NWP = C * C;                 // 4,194,304
  const int NHB = 2 * 16 * 2048 * 128;   // per q/k/v buffer = 8,388,608

  u16* xb  = (u16*)d_ws;
  u16* wat = xb  + NX;    // W_attn^T bf16 [6144][2048]
  u16* wpt = wat + NWA;   // W_proj^T bf16 [2048][2048]
  u16* qb  = wpt + NWP;   // Q  [B,H,T,HD]
  u16* kbf = qb  + NHB;   // K  [B,H,T,HD]
  u16* vbf = kbf + NHB;   // V^T [B,H,HD,T]
  u16* yb  = vbf + NHB;

  // 1) conversions (+ weight transposes so GEMM B tiles are async-DMA friendly)
  cvt_f32_bf16<<<(NX + 255) / 256, 256, 0, stream>>>(x, xb, NX);
  cvt_transpose_bf16<<<(NWA + 255) / 256, 256, 0, stream>>>(W_attn, wat, C, N3);
  cvt_transpose_bf16<<<(NWP + 255) / 256, 256, 0, stream>>>(W_proj, wpt, C, C);

  // 2) QKV GEMM (bias fused; Q/K per-head, V per-head transposed)
  gemm_bf16_kernel<<<dim3(N3 / 128, M / 128), 256, 0, stream>>>(
      xb, wat, b_attn, M, N3, C, /*mode=*/0, qb, kbf, vbf, nullptr);

  // 3) RoPE on Q (fused 1/sqrt(HD)) and K
  const int npairs = 2 * 16 * 2048 * 64;
  const float scale = 0.08838834764831845f;  // 1/sqrt(128)
  rope_kernel<<<(npairs + 255) / 256, 256, 0, stream>>>(qb,  cosp, sinp, scale, npairs);
  rope_kernel<<<(npairs + 255) / 256, 256, 0, stream>>>(kbf, cosp, sinp, 1.0f,  npairs);

  // 4) causal flash attention -> bf16 y [4096,2048]
  flash_attn_kernel<<<1024, 128, 0, stream>>>(qb, kbf, vbf, yb);

  // 5) output projection -> f32 out
  gemm_bf16_kernel<<<dim3(C / 128, M / 128), 256, 0, stream>>>(
      yb, wpt, b_proj, M, C, C, /*mode=*/1, nullptr, nullptr, nullptr, out);
}
